// GIFStream2DCodecEvaluator_24352464570198
// MI455X (gfx1250) — matrix operand order repaired and checked by reading, not compiled, not run on hardware
//
#include <hip/hip_runtime.h>
#include <math.h>

typedef __attribute__((ext_vector_type(16))) _Float16 v16h;
typedef __attribute__((ext_vector_type(8)))  float    v8f;

#define WAVES_PER_BLOCK 4

// ---- packed-weight layout inside d_ws (all f16, K padded to 64, column-major WT[n][64]) ----
#define OFF_W1CAT  0        // 3 x [64][64]  (op_w1, col_w1, cov_w1)
#define OFF_OPW2   12288    // [16][64]
#define OFF_COLW2  13312    // [32][64]
#define OFF_COVW2  15360    // [80][64]
#define OFF_MOTW1  20480    // [64][64]
#define OFF_MOTW2  24576    // [48][64]
#define WPK_HALVES 27648
// biases (f32, zero padded) appended after the f16 weights
#define BOFF_B1    0        // 3 x 64 (op_b1, col_b1, cov_b1)
#define BOFF_OPB2  192      // 16
#define BOFF_COLB2 208      // 32
#define BOFF_COVB2 240      // 80
#define BOFF_MOTB1 320      // 64
#define BOFF_MOTB2 384      // 48

__device__ __forceinline__ unsigned pack2h(float a, float b) {
  union { unsigned u; _Float16 h[2]; } r;
  r.h[0] = (_Float16)a; r.h[1] = (_Float16)b;
  return r.u;
}

// ---------------- weight prepack: f32 row-major -> f16 transposed, zero-padded ----------------
__global__ void prepack_weights(const float* __restrict__ op_w1, const float* __restrict__ col_w1,
                                const float* __restrict__ cov_w1, const float* __restrict__ op_w2,
                                const float* __restrict__ col_w2, const float* __restrict__ cov_w2,
                                const float* __restrict__ mot_w1, const float* __restrict__ mot_w2,
                                const float* __restrict__ op_b1,  const float* __restrict__ col_b1,
                                const float* __restrict__ cov_b1, const float* __restrict__ op_b2,
                                const float* __restrict__ col_b2, const float* __restrict__ cov_b2,
                                const float* __restrict__ mot_b1, const float* __restrict__ mot_b2,
                                _Float16* __restrict__ wpk) {
  const int t      = threadIdx.x + blockIdx.x * blockDim.x;
  const int stride = blockDim.x * gridDim.x;
  float* bpk = reinterpret_cast<float*>(wpk + WPK_HALVES);

  auto packW = [&](const float* W, int Krows, int Ncols, int Npad, int base) {
    const int total = Npad * 64;
    for (int i = t; i < total; i += stride) {
      const int n = i >> 6, k = i & 63;
      const float x = (k < Krows && n < Ncols) ? W[k * Ncols + n] : 0.0f;
      wpk[base + i] = (_Float16)x;
    }
  };
  auto packB = [&](const float* B, int Ncols, int Npad, int fbase) {
    for (int i = t; i < Npad; i += stride) bpk[fbase + i] = (i < Ncols) ? B[i] : 0.0f;
  };

  packW(op_w1,  51, 64, 64, OFF_W1CAT);
  packW(col_w1, 51, 64, 64, OFF_W1CAT + 4096);
  packW(cov_w1, 51, 64, 64, OFF_W1CAT + 8192);
  packW(op_w2,  64, 10, 16, OFF_OPW2);
  packW(col_w2, 64, 30, 32, OFF_COLW2);
  packW(cov_w2, 64, 70, 80, OFF_COVW2);
  packW(mot_w1, 64, 64, 64, OFF_MOTW1);
  packW(mot_w2, 64, 37, 48, OFF_MOTW2);
  packB(op_b1,  64, 64, BOFF_B1);
  packB(col_b1, 64, 64, BOFF_B1 + 64);
  packB(cov_b1, 64, 64, BOFF_B1 + 128);
  packB(op_b2,  10, 16, BOFF_OPB2);
  packB(col_b2, 30, 32, BOFF_COLB2);
  packB(cov_b2, 70, 80, BOFF_COVB2);
  packB(mot_b1, 64, 64, BOFF_MOTB1);
  packB(mot_b2, 37, 48, BOFF_MOTB2);
}

// ---- A fragment from LDS tile (row-major, ldh halves/row), CDNA5 16-bit A 16x32 layout ----
__device__ __forceinline__ v16h lds_load_a_frag(const _Float16* base, int ldh, int ks, int lane) {
  const int row = lane & 15;
  const int hi  = lane >> 4;
  union { v16h v; unsigned u[8]; } r;
#pragma unroll
  for (int vv = 0; vv < 8; ++vv) {
    const int k = ks * 32 + ((vv < 4) ? 0 : 16) + hi * 8 + ((vv & 3) << 1);
    r.u[vv] = *reinterpret_cast<const unsigned*>(base + row * ldh + k);
  }
  return r.v;
}

// ---- B fragment from packed transposed f16 weights: 2 contiguous b128 loads per lane ----
__device__ __forceinline__ v16h ldg_b_packed(const _Float16* WT, int nt, int ks, int lane) {
  const int col = lane & 15;
  const int hi  = lane >> 4;
  const _Float16* p = WT + (((nt << 4) + col) << 6) + ks * 32 + hi * 8;
  union { v16h v; uint4 q[2]; } r;
  r.q[0] = *reinterpret_cast<const uint4*>(p);
  r.q[1] = *reinterpret_cast<const uint4*>(p + 16);
  return r.v;
}

// D = A(16x64) * WT[tile nt]; accumulator starts at inline 0, bias applied by caller.
__device__ __forceinline__ v8f wmma_k64(v16h a0, v16h a1, const _Float16* WT, int nt, int lane) {
  v8f c = { 0.f, 0.f, 0.f, 0.f, 0.f, 0.f, 0.f, 0.f };
  c = __builtin_amdgcn_wmma_f32_16x16x32_f16(false, a0, false, ldg_b_packed(WT, nt, 0, lane),
                                             (short)0, c, false, false);
  c = __builtin_amdgcn_wmma_f32_16x16x32_f16(false, a1, false, ldg_b_packed(WT, nt, 1, lane),
                                             (short)0, c, false, false);
  return c;
}

__global__ __launch_bounds__(128) void gifstream_fused_kernel(
    const float* __restrict__ cam,     const float* __restrict__ anchors,
    const float* __restrict__ scales,  const float* __restrict__ feat,
    const float* __restrict__ timef,   const float* __restrict__ factors,
    const int*   __restrict__ vis,     const int*   __restrict__ knn,
    const _Float16* __restrict__ wpk,
    float* __restrict__ out, int M)
{
  __shared__ _Float16 s_taf [WAVES_PER_BLOCK][16][64];          // main MLP input (51 used)
  __shared__ _Float16 s_tafm[WAVES_PER_BLOCK][16][64];          // motion MLP input (64)
  __shared__ _Float16 s_hid [WAVES_PER_BLOCK][16][64];          // one hidden layer at a time
  __shared__ __align__(16) float s_stage[WAVES_PER_BLOCK][1760]; // out_m (800) then out_mk (1760)
  __shared__ float s_pf[WAVES_PER_BLOCK][16];
  __shared__ float s_mf[WAVES_PER_BLOCK][16];
  __shared__ float s_kf[WAVES_PER_BLOCK][16];

  const int tid      = threadIdx.x;
  const int w        = tid >> 5;
  const int lane     = tid & 31;
  const int row      = lane & 15;
  const int hi       = lane >> 4;
  const int colN     = lane & 15;
  const int tileBase = blockIdx.x * (WAVES_PER_BLOCK * 16) + w * 16;
  const int m        = tileBase + row;

  const float* biaspk = reinterpret_cast<const float*>(wpk + WPK_HALVES);
  float* stage = &s_stage[w][0];

  // ---------------- Phase 1: gathers + input assembly ----------------
  float accF[32];
  float accT[16];
  if (m < M) {
    const int idx = vis[m];
    if (lane < 16) {
      const float tff = factors[idx * 4 + 0];
      const float mfv = factors[idx * 4 + 1];
      const float kfv = factors[idx * 4 + 2];
      const float pfv = factors[idx * 4 + 3];
      s_pf[w][row] = pfv; s_mf[w][row] = mfv; s_kf[w][row] = kfv;

      const float4* fp4 = reinterpret_cast<const float4*>(feat + (size_t)idx * 32);
      unsigned* trow = reinterpret_cast<unsigned*>(&s_taf[w][row][0]);
#pragma unroll
      for (int q = 0; q < 8; ++q) {
        const float4 f = fp4[q];
        trow[2 * q + 0] = pack2h(f.x, f.y);
        trow[2 * q + 1] = pack2h(f.z, f.w);
      }

      const float cx = cam[3], cy = cam[7], cz = cam[11];
      const float ax = anchors[idx * 3 + 0], ay = anchors[idx * 3 + 1], az = anchors[idx * 3 + 2];
      const float vx = ax - cx, vy = ay - cy, vz = az - cz;
      const float inv = 1.0f / fmaxf(sqrtf(vx * vx + vy * vy + vz * vz), 1e-8f);
      s_taf[w][row][32] = (_Float16)(vx * inv);
      s_taf[w][row][33] = (_Float16)(vy * inv);
      s_taf[w][row][34] = (_Float16)(vz * inv);

      const float4* tp4 = reinterpret_cast<const float4*>(timef + (size_t)idx * 256 + 112);
      float tfv[16];
#pragma unroll
      for (int q = 0; q < 4; ++q) {
        const float4 tq = tp4[q];
        tfv[4 * q + 0] = tq.x; tfv[4 * q + 1] = tq.y;
        tfv[4 * q + 2] = tq.z; tfv[4 * q + 3] = tq.w;
      }
#pragma unroll
      for (int c = 0; c < 16; ++c) s_taf[w][row][35 + c] = (_Float16)(tfv[c] * tff);
#pragma unroll
      for (int d = 51; d < 64; ++d) s_taf[w][row][d] = (_Float16)0.0f;

      // out_m tail -> stage[row*50 + 37..49]: [scales(6) | factors(4) | anchor(3)]
      float* sm = stage + row * 50;
#pragma unroll
      for (int i = 0; i < 3; ++i) sm[37 + i] = expf(scales[idx * 6 + i]);
#pragma unroll
      for (int i = 0; i < 3; ++i) sm[40 + i] = expf(scales[idx * 6 + 3 + i]) * pfv;
      sm[43] = tff; sm[44] = mfv; sm[45] = kfv; sm[46] = pfv;
      sm[47] = ax;  sm[48] = ay;  sm[49] = az;
    } else {
#pragma unroll
      for (int d = 0; d < 32; ++d) accF[d] = 0.0f;
#pragma unroll
      for (int c = 0; c < 16; ++c) accT[c] = 0.0f;
      for (int kn = 0; kn < 6; ++kn) {
        const int nb = knn[idx * 6 + kn];
        const float4* fp4 = reinterpret_cast<const float4*>(feat + (size_t)nb * 32);
#pragma unroll
        for (int q = 0; q < 8; ++q) {
          const float4 f = fp4[q];
          accF[4 * q + 0] += f.x; accF[4 * q + 1] += f.y;
          accF[4 * q + 2] += f.z; accF[4 * q + 3] += f.w;
        }
        const float4* tp4 = reinterpret_cast<const float4*>(timef + (size_t)nb * 256 + 112);
#pragma unroll
        for (int q = 0; q < 4; ++q) {
          const float4 tq = tp4[q];
          accT[4 * q + 0] += tq.x; accT[4 * q + 1] += tq.y;
          accT[4 * q + 2] += tq.z; accT[4 * q + 3] += tq.w;
        }
      }
    }
  }
  __syncthreads();

  if (m < M && lane >= 16) {
    const float kfv = s_kf[w][row];
    const float okf = (1.0f - kfv) * (1.0f / 6.0f);
    unsigned* mrow = reinterpret_cast<unsigned*>(&s_tafm[w][row][0]);
#pragma unroll
    for (int d = 0; d < 32; d += 2) {
      const float b0 = (float)s_taf[w][row][d];
      const float b1 = (float)s_taf[w][row][d + 1];
      mrow[d >> 1] = pack2h(kfv * b0 + okf * accF[d], kfv * b1 + okf * accF[d + 1]);
    }
#pragma unroll
    for (int c = 0; c < 16; c += 2) {
      const float b0 = (float)s_taf[w][row][35 + c];
      const float b1 = (float)s_taf[w][row][36 + c];
      mrow[(32 + c) >> 1] = pack2h(kfv * b0 + okf * accT[c], kfv * b1 + okf * accT[c + 1]);
    }
#pragma unroll
    for (int n = 0; n < 8; n += 2) {   // temb: sin/cos(2^n * pi * 0.5)
      const float a0 = (float)(1 << n)       * 1.57079632679489662f;
      const float a1 = (float)(1 << (n + 1)) * 1.57079632679489662f;
      mrow[(48 + n) >> 1] = pack2h(sinf(a0), sinf(a1));
      mrow[(56 + n) >> 1] = pack2h(cosf(a0), cosf(a1));
    }
  }
  __syncthreads();

  // ---------------- motion MLP layer 1: taf_ @ mot_w1 -> s_hid ----------------
  {
    const v16h a0 = lds_load_a_frag(&s_tafm[w][0][0], 64, 0, lane);
    const v16h a1 = lds_load_a_frag(&s_tafm[w][0][0], 64, 1, lane);
#pragma unroll
    for (int nt = 0; nt < 4; ++nt) {
      v8f c = wmma_k64(a0, a1, wpk + OFF_MOTW1, nt, lane);
      const float b = biaspk[BOFF_MOTB1 + nt * 16 + colN];
#pragma unroll
      for (int r = 0; r < 8; ++r)
        s_hid[w][r + 8 * hi][nt * 16 + colN] = (_Float16)fmaxf(c[r] + b, 0.0f);
    }
  }
  __syncthreads();

  // ---------------- motion MLP layer 2 -> stage[ml*50 + 0..36] ----------------
  {
    const v16h h0 = lds_load_a_frag(&s_hid[w][0][0], 64, 0, lane);
    const v16h h1 = lds_load_a_frag(&s_hid[w][0][0], 64, 1, lane);
#pragma unroll
    for (int nt = 0; nt < 3; ++nt) {
      v8f c = wmma_k64(h0, h1, wpk + OFF_MOTW2, nt, lane);
      const int cg = nt * 16 + colN;
      if (cg < 37) {
        const float b = biaspk[BOFF_MOTB2 + cg];
#pragma unroll
        for (int r = 0; r < 8; ++r) {
          const int ml = r + 8 * hi;
          stage[ml * 50 + cg] = (c[r] + b) * s_mf[w][ml];
        }
      }
    }
  }
  __syncthreads();

  // ---------------- coalesced store of out_m tile (800 contiguous floats) ----------------
  {
    float* out_m = out + (size_t)M * 110;
    if (tileBase + 16 <= M) {
      const uint4* s4 = reinterpret_cast<const uint4*>(stage);
      uint4* d4 = reinterpret_cast<uint4*>(out_m + (size_t)tileBase * 50);
      for (int i = lane; i < 200; i += 32) d4[i] = s4[i];
    } else if (tileBase < M) {
      const int nelem = (M - tileBase) * 50;
      float* d = out_m + (size_t)tileBase * 50;
      for (int i = lane; i < nelem; i += 32) d[i] = stage[i];
    }
  }
  __syncthreads();  // stage free for reuse by out_mk

  // ---------------- main MLPs, one at a time: layer1 -> s_hid -> layer2 -> stage ----------------
  {
    const v16h a0 = lds_load_a_frag(&s_taf[w][0][0], 64, 0, lane);
    const v16h a1 = lds_load_a_frag(&s_taf[w][0][0], 64, 1, lane);
    for (int p = 0; p < 3; ++p) {
      // layer 1
      const _Float16* W1 = wpk + OFF_W1CAT + p * 4096;
#pragma unroll
      for (int nt = 0; nt < 4; ++nt) {
        v8f c = wmma_k64(a0, a1, W1, nt, lane);
        const float b = biaspk[BOFF_B1 + p * 64 + nt * 16 + colN];
#pragma unroll
        for (int r = 0; r < 8; ++r)
          s_hid[w][r + 8 * hi][nt * 16 + colN] = (_Float16)fmaxf(c[r] + b, 0.0f);
      }
      __syncthreads();

      // layer 2 + activation -> stage (disjoint columns per p)
      const v16h h0 = lds_load_a_frag(&s_hid[w][0][0], 64, 0, lane);
      const v16h h1 = lds_load_a_frag(&s_hid[w][0][0], 64, 1, lane);
      if (p == 0) {           // opacity: tanh(. ) * pf -> col 0 of 11
        v8f c = wmma_k64(h0, h1, wpk + OFF_OPW2, 0, lane);
        if (colN < 10) {
          const float b = biaspk[BOFF_OPB2 + colN];
#pragma unroll
          for (int r = 0; r < 8; ++r) {
            const int ml = r + 8 * hi;
            stage[(ml * 10 + colN) * 11] = tanhf(c[r] + b) * s_pf[w][ml];
          }
        }
      } else if (p == 1) {    // color: sigmoid -> cols 1..3
#pragma unroll
        for (int nt = 0; nt < 2; ++nt) {
          v8f c = wmma_k64(h0, h1, wpk + OFF_COLW2, nt, lane);
          const int cg = nt * 16 + colN;
          if (cg < 30) {
            const float b = biaspk[BOFF_COLB2 + cg];
            const int j = cg / 3, ch = cg % 3;
#pragma unroll
            for (int r = 0; r < 8; ++r) {
              const int ml = r + 8 * hi;
              stage[(ml * 10 + j) * 11 + 1 + ch] = 1.0f / (1.0f + expf(-(c[r] + b)));
            }
          }
        }
      } else {                // cov/sr: identity -> cols 4..10
#pragma unroll
        for (int nt = 0; nt < 5; ++nt) {
          v8f c = wmma_k64(h0, h1, wpk + OFF_COVW2, nt, lane);
          const int cg = nt * 16 + colN;
          if (cg < 70) {
            const float b = biaspk[BOFF_COVB2 + cg];
            const int j = cg / 7, ch = cg % 7;
#pragma unroll
            for (int r = 0; r < 8; ++r) {
              const int ml = r + 8 * hi;
              stage[(ml * 10 + j) * 11 + 4 + ch] = c[r] + b;
            }
          }
        }
      }
      __syncthreads();  // hid reuse by next p / stage complete before store
    }
  }

  // ---------------- coalesced store of out_mk tile (1760 contiguous floats) ----------------
  {
    if (tileBase + 16 <= M) {
      const uint4* s4 = reinterpret_cast<const uint4*>(stage);
      uint4* d4 = reinterpret_cast<uint4*>(out + (size_t)tileBase * 110);
      for (int i = lane; i < 440; i += 32) d4[i] = s4[i];
    } else if (tileBase < M) {
      const int nelem = (M - tileBase) * 110;
      float* d = out + (size_t)tileBase * 110;
      for (int i = lane; i < nelem; i += 32) d[i] = stage[i];
    }
  }
}

extern "C" void kernel_launch(void* const* d_in, const int* in_sizes, int n_in,
                              void* d_out, int out_size, void* d_ws, size_t ws_size,
                              hipStream_t stream) {
  (void)n_in; (void)out_size; (void)ws_size;
  const float* cam     = (const float*)d_in[0];
  const float* anchors = (const float*)d_in[1];
  const float* scales  = (const float*)d_in[2];
  const float* feat    = (const float*)d_in[3];
  const float* timef   = (const float*)d_in[4];
  const float* factors = (const float*)d_in[5];
  const int*   vis     = (const int*)d_in[22];
  const int*   knn     = (const int*)d_in[23];
  _Float16*    wpk     = (_Float16*)d_ws;   // 27648 f16 + 432 f32 = ~57 KB

  prepack_weights<<<32, 256, 0, stream>>>(
      (const float*)d_in[6],  (const float*)d_in[10], (const float*)d_in[14],
      (const float*)d_in[8],  (const float*)d_in[12], (const float*)d_in[16],
      (const float*)d_in[18], (const float*)d_in[20],
      (const float*)d_in[7],  (const float*)d_in[11], (const float*)d_in[15],
      (const float*)d_in[9],  (const float*)d_in[13], (const float*)d_in[17],
      (const float*)d_in[19], (const float*)d_in[21], wpk);

  const int M = in_sizes[22];
  const int rowsPerBlock = WAVES_PER_BLOCK * 16;  // 64
  const int blocks = (M + rowsPerBlock - 1) / rowsPerBlock;

  gifstream_fused_kernel<<<blocks, 128, 0, stream>>>(
      cam, anchors, scales, feat, timef, factors,
      vis, knn, wpk, (float*)d_out, M);
}
